// RSSM_28570122453419
// MI455X (gfx1250) — compile-verified, matrix operand order checked by reading, not run on hardware
//
#include <hip/hip_runtime.h>
#include <stdint.h>

// ---------------- problem constants ----------------
#define BB     16
#define LL     64
#define DETER  4096
#define HIDDEN 1024
#define BLOCKS 8
#define DBLK   512       // DETER / BLOCKS
#define FLAT   1024      // STOCH * DISC
#define EMB    1024
#define DISC   32
#define STOCH  32

#define NBLK_GRID 64     // persistent workgroups (must all be co-resident)
#define THREADS   256    // 8 wave32 waves per workgroup

// output tuple layout: (stochs, deters, logits) concatenated flat, fp32
#define O_ST  ((size_t)0)
#define O_DE  (O_ST + (size_t)BB * LL * FLAT)
#define O_LG  (O_DE + (size_t)BB * LL * DETER)

// packed bf16 weight region sizes in dwords: (N/16 tiles) * (K/32 chunks) * 256
#define PW_IN0 ((size_t)64 * 128 * 256)
#define PW_IN1 ((size_t)64 * 32  * 256)
#define PW_IN2 ((size_t)64 * 1   * 256)
#define PW_IN3 ((size_t)64 * 1   * 256)   // K padded 16 -> 32 with zeros
#define PW_HID ((size_t)8  * 32  * 144 * 256)
#define PW_GRU ((size_t)8  * 96  * 16  * 256)
#define PW_OBS ((size_t)64 * 160 * 256)
#define PW_LW  ((size_t)64 * 32  * 256)

typedef __attribute__((ext_vector_type(16))) __bf16   v16bf;
typedef __attribute__((ext_vector_type(8)))  float    v8f;
typedef __attribute__((ext_vector_type(4)))  unsigned u32x4;
typedef __attribute__((ext_vector_type(8)))  unsigned u32x8;

// ---------------- small device helpers ----------------
__device__ __forceinline__ unsigned short f2bf(float f) {
  unsigned u = __float_as_uint(f);
  unsigned r = ((u >> 16) & 1u) + 0x7fffu;     // round-to-nearest-even
  return (unsigned short)((u + r) >> 16);
}
__device__ __forceinline__ float sigm(float x) { return 1.f / (1.f + __expf(-x)); }
__device__ __forceinline__ float siluf(float x) { return x / (1.f + __expf(-x)); }

__device__ __forceinline__ v16bf comb16(u32x4 a, u32x4 b) {
  u32x8 d = __builtin_shufflevector(a, b, 0, 1, 2, 3, 4, 5, 6, 7);
  return __builtin_bit_cast(v16bf, d);
}

// A operand (16 x K row-major bf16).  CDNA5 16-bit A layout:
// lanes 0-15: M=lane, K = {k0..k0+7, k0+16..k0+23}
// lanes 16-31: M=lane-16, K = {k0+8..k0+15, k0+24..k0+31}
__device__ __forceinline__ v16bf load_a16(const unsigned short* A, int lda, int lane, int k0) {
  int m  = lane & 15;
  int kb = (lane & 16) ? 8 : 0;
  const unsigned short* p = A + (size_t)m * lda + k0 + kb;
  u32x4 q0 = *(const u32x4*)(p);
  u32x4 q1 = *(const u32x4*)(p + 16);
  return comb16(q0, q1);
}

// B operand pre-packed by pack_bf16_B: per (ntile,kchunk) 1024B block,
// lane reads 8 consecutive dwords at +lane*32B (fully coalesced).
__device__ __forceinline__ v16bf load_b16(const unsigned* Btile, int lane, int kc) {
  const u32x4* p = (const u32x4*)(Btile + kc * 256 + lane * 8);
  return comb16(p[0], p[1]);
}

__device__ __forceinline__ v8f wmma_bf16(v16bf a, v16bf b, v8f c) {
  return __builtin_amdgcn_wmma_f32_16x16x32_bf16(false, a, false, b, (short)0, c, false, false);
}

__device__ __forceinline__ float wave_sum(float v) {
#pragma unroll
  for (int o = 16; o; o >>= 1) v += __shfl_xor(v, o, 32);
  return v;
}

// device-wide barrier: monotonic epoch counter, all NBLK_GRID blocks resident
__device__ __forceinline__ void gbar(unsigned* cnt, unsigned& epoch, int nb) {
  ++epoch;
  __threadfence();
  __syncthreads();
  if (threadIdx.x == 0) {
    __hip_atomic_fetch_add(cnt, 1u, __ATOMIC_ACQ_REL, __HIP_MEMORY_SCOPE_AGENT);
    while (__hip_atomic_load(cnt, __ATOMIC_ACQUIRE, __HIP_MEMORY_SCOPE_AGENT) <
           epoch * (unsigned)nb)
      __builtin_amdgcn_s_sleep(2);
  }
  __syncthreads();
}

// ---------------- prep: pack fp32 W[K,N] (row-major) -> WMMA-B bf16 tile layout ----
__global__ void pack_bf16_B(const float* __restrict__ W, unsigned* __restrict__ dst,
                            int Kpad, int Ksrc, int N) {
  int blk = blockIdx.y;                       // for blocked weights (hid_w, gru_w)
  const float* Wb = W + (size_t)blk * Ksrc * N;
  int kch = Kpad / 32, ntiles = N / 16;
  size_t total = (size_t)ntiles * kch * 256;
  unsigned* Db = dst + (size_t)blk * total;
  for (size_t i = (size_t)blockIdx.x * blockDim.x + threadIdx.x; i < total;
       i += (size_t)gridDim.x * blockDim.x) {
    int r = (int)(i & 7);
    int lane = (int)((i >> 3) & 31);
    size_t rem = i >> 8;
    int kc = (int)(rem % kch);
    int nt = (int)(rem / kch);
    int n  = nt * 16 + (lane & 15);
    int kb = (lane & 16) ? 8 : 0;
    int klo = kb + ((r < 4) ? (2 * r) : (16 + 2 * (r - 4)));
    int k0 = kc * 32 + klo;
    float f0 = (k0     < Ksrc) ? Wb[(size_t)k0 * N + n]       : 0.f;
    float f1 = (k0 + 1 < Ksrc) ? Wb[(size_t)(k0 + 1) * N + n] : 0.f;
    Db[i] = (unsigned)f2bf(f0) | ((unsigned)f2bf(f1) << 16);
  }
}

__global__ void prep_misc(const float* __restrict__ embed, const float* __restrict__ stoch0,
                          const float* __restrict__ deter0, unsigned short* __restrict__ emb_bf,
                          float* __restrict__ sto, float* __restrict__ det,
                          unsigned* __restrict__ bar) {
  size_t tid = (size_t)blockIdx.x * blockDim.x + threadIdx.x;
  size_t NT  = (size_t)gridDim.x * blockDim.x;
  if (tid == 0) bar[0] = 0u;
  for (size_t i = tid; i < (size_t)BB * LL * EMB; i += NT) emb_bf[i] = f2bf(embed[i]);
  for (size_t i = tid; i < (size_t)BB * FLAT; i += NT) sto[i] = stoch0[i];
  for (size_t i = tid; i < (size_t)BB * DETER; i += NT) det[i] = deter0[i];
}

// ---------------- persistent scan kernel ----------------
struct RP {
  // inputs
  const float *action, *d_emb;
  const unsigned char* reset;                 // jnp bool -> 1 byte
  // biases / gains
  const float *in0_b, *in0_g, *in1_b, *in1_g, *in2_b, *in2_g, *in3_b, *in3_g;
  const float *hid_b, *hid_g, *gru_b, *obs_b0, *obs_g0, *obs_lb;
  // packed bf16 weights (WMMA-B layout)
  const unsigned *w_in0, *w_in1, *w_in2, *w_in3, *w_hid, *w_gru, *w_obs, *w_lw;
  // bf16 activation buffers
  const unsigned short* emb_bf;
  unsigned short *A0, *A1, *A2, *A3, *xbf, *hbf, *ndbf, *albf;
  // f32 buffers
  float *xpre, *hpre, *opre, *logit, *det, *sto, *msx, *msh, *mso;
  unsigned* bar;
  float* out;
};

__global__ void __launch_bounds__(THREADS) rssm_scan_persistent(RP p) {
  const int NB   = gridDim.x;
  const int tid  = blockIdx.x * blockDim.x + threadIdx.x;
  const int NT   = gridDim.x * blockDim.x;
  const int lane = threadIdx.x & 31;
  const int gw   = blockIdx.x * (blockDim.x >> 5) + (threadIdx.x >> 5);
  const int GW   = NB * (blockDim.x >> 5);
  unsigned epoch = 0;

  for (int t = 0; t < LL; ++t) {
    // ---- E0: reset-mask state, build bf16 A operands, zero x accumulator ----
    for (int i = tid; i < BB * DETER; i += NT) {
      int b = i >> 12;
      float v = p.det[i];
      if (p.reset[b * LL + t]) v = 0.f;
      p.det[i] = v;
      p.A0[i] = f2bf(v);
      p.xpre[i] = 0.f;
    }
    for (int i = tid; i < BB * FLAT; i += NT) {
      int b = i >> 10;
      float v = p.sto[i];
      if (p.reset[b * LL + t]) v = 0.f;
      p.sto[i] = v;
      p.A1[i] = f2bf(v);
    }
    for (int i = tid; i < BB * 32; i += NT) {
      int b = i >> 5, j = i & 31;
      float a = p.action[((size_t)b * LL + t) * 32 + j];
      a = a / fmaxf(fabsf(a), 1.f);
      if (p.reset[b * LL + t]) a = 0.f;
      p.A2[i] = f2bf(a);
    }
    for (int i = tid; i < BB * 32; i += NT) {
      int b = i >> 5, j = i & 31;
      float v = (j < 16) ? p.d_emb[((size_t)b * LL + t) * 16 + j] : 0.f;
      p.A3[i] = f2bf(v);
    }
    gbar(p.bar, epoch, NB);

    // ---- G1: four input GEMMs into xpre (K-split + f32 atomics for x0) ----
    for (int task = gw; task < 448; task += GW) {
      const unsigned short* A;
      const unsigned* Wp;
      int lda, nt, kc0, kcN, kcTot, segBase;
      if (task < 256) { nt = task >> 2; int ks = task & 3; A = p.A0; lda = DETER;
                        Wp = p.w_in0; kcTot = 128; kc0 = ks * 32; kcN = 32; segBase = 0; }
      else if (task < 320) { nt = task - 256; A = p.A1; lda = FLAT;
                        Wp = p.w_in1; kcTot = 32; kc0 = 0; kcN = 32; segBase = 1024; }
      else if (task < 384) { nt = task - 320; A = p.A2; lda = 32;
                        Wp = p.w_in2; kcTot = 1; kc0 = 0; kcN = 1; segBase = 2048; }
      else { nt = task - 384; A = p.A3; lda = 32;
                        Wp = p.w_in3; kcTot = 1; kc0 = 0; kcN = 1; segBase = 3072; }
      const unsigned* Btile = Wp + (size_t)nt * kcTot * 256;
      v8f acc = {};
      for (int kc = kc0; kc < kc0 + kcN; ++kc)
        acc = wmma_bf16(load_a16(A, lda, lane, kc * 32), load_b16(Btile, lane, kc), acc);
      int n = segBase + nt * 16 + (lane & 15);
      int mb = (lane & 16) ? 8 : 0;
#pragma unroll
      for (int r = 0; r < 8; r++) atomicAdd(&p.xpre[(size_t)(mb + r) * DETER + n], acc[r]);
    }
    gbar(p.bar, epoch, NB);

    // ---- E2a: sum-of-squares per (row, 1024-segment) ----
    for (int w = gw; w < 64; w += GW) {
      int b = w >> 2, seg = w & 3;
      const float* bias = (seg == 0) ? p.in0_b : (seg == 1) ? p.in1_b
                        : (seg == 2) ? p.in2_b : p.in3_b;
      float s = 0.f;
      for (int i = lane; i < 1024; i += 32) {
        float v = p.xpre[(size_t)b * DETER + seg * 1024 + i] + bias[i];
        s += v * v;
      }
      s = wave_sum(s);
      if (lane == 0) p.msx[w] = s * (1.f / 1024.f);
    }
    gbar(p.bar, epoch, NB);

    // ---- E2b: rmsnorm + SiLU -> xbf; zero hpre ----
    for (int i = tid; i < BB * DETER; i += NT) {
      int b = i >> 12, c = i & 4095, seg = c >> 10, cs = c & 1023;
      const float* bias = (seg == 0) ? p.in0_b : (seg == 1) ? p.in1_b
                        : (seg == 2) ? p.in2_b : p.in3_b;
      const float* g = (seg == 0) ? p.in0_g : (seg == 1) ? p.in1_g
                     : (seg == 2) ? p.in2_g : p.in3_g;
      float v = (p.xpre[i] + bias[cs]) * rsqrtf(p.msx[b * 4 + seg] + 1e-4f) * g[cs];
      p.xbf[i] = f2bf(siluf(v));
      p.hpre[i] = 0.f;
    }
    gbar(p.bar, epoch, NB);

    // ---- G3: block hidden GEMM, A = [deter_blk(512) | x(4096)], K=4608, split 2 ----
    for (int task = gw; task < 512; task += GW) {
      int ks = task & 1, nt = (task >> 1) & 31, blk = task >> 6;
      const unsigned* Btile = p.w_hid + ((size_t)blk * 32 + nt) * 144 * 256;
      v8f acc = {};
      int kc0 = ks * 72;
      for (int kc = kc0; kc < kc0 + 72; ++kc) {
        if ((kc & 15) == 0 && kc + 16 < 144)
          __builtin_prefetch(Btile + (kc + 16) * 256, 0, 0);
        const unsigned short* A;
        int k;
        if (kc < 16) { A = p.A0 + blk * DBLK; k = kc * 32; }
        else         { A = p.xbf;             k = (kc - 16) * 32; }
        acc = wmma_bf16(load_a16(A, DETER, lane, k), load_b16(Btile, lane, kc), acc);
      }
      int n = blk * DBLK + nt * 16 + (lane & 15);
      int mb = (lane & 16) ? 8 : 0;
#pragma unroll
      for (int r = 0; r < 8; r++) atomicAdd(&p.hpre[(size_t)(mb + r) * DETER + n], acc[r]);
    }
    gbar(p.bar, epoch, NB);

    // ---- E4a: sumsq per row over 4096 ----
    for (int w = gw; w < BB; w += GW) {
      float s = 0.f;
      for (int i = lane; i < DETER; i += 32) {
        float v = p.hpre[(size_t)w * DETER + i] + p.hid_b[i];
        s += v * v;
      }
      s = wave_sum(s);
      if (lane == 0) p.msh[w] = s * (1.f / 4096.f);
    }
    gbar(p.bar, epoch, NB);

    // ---- E4b: rmsnorm + SiLU -> hbf; zero opre ----
    for (int i = tid; i < BB * DETER; i += NT) {
      int b = i >> 12, c = i & 4095;
      float v = (p.hpre[i] + p.hid_b[c]) * rsqrtf(p.msh[b] + 1e-4f) * p.hid_g[c];
      p.hbf[i] = f2bf(siluf(v));
    }
    for (int i = tid; i < BB * HIDDEN; i += NT) p.opre[i] = 0.f;
    gbar(p.bar, epoch, NB);

    // ---- G5: GRU block GEMM (3 gate tiles per task) fused with gate math ----
    for (int task = gw; task < 256; task += GW) {
      int ct = task & 31, blk = task >> 5;
      const unsigned* Bblk = p.w_gru + (size_t)blk * 96 * 16 * 256;
      const unsigned short* A = p.hbf + blk * DBLK;
      v8f ar = {}, ac = {}, au = {};
      for (int kc = 0; kc < 16; ++kc) {
        v16bf a = load_a16(A, DETER, lane, kc * 32);
        ar = wmma_bf16(a, load_b16(Bblk + (size_t)(ct)      * 16 * 256, lane, kc), ar);
        ac = wmma_bf16(a, load_b16(Bblk + (size_t)(32 + ct) * 16 * 256, lane, kc), ac);
        au = wmma_bf16(a, load_b16(Bblk + (size_t)(64 + ct) * 16 * 256, lane, kc), au);
      }
      int nin = ct * 16 + (lane & 15);
      int mb = (lane & 16) ? 8 : 0;
#pragma unroll
      for (int r = 0; r < 8; r++) {
        int m = mb + r;
        float rg = sigm(ar[r] + p.gru_b[blk * 1536 + nin]);
        float cc = tanhf(rg * (ac[r] + p.gru_b[blk * 1536 + 512 + nin]));
        float ug = sigm(au[r] + p.gru_b[blk * 1536 + 1024 + nin] - 1.f);
        int col = blk * DBLK + nin;
        float prev = p.det[(size_t)m * DETER + col];
        float nd = ug * cc + (1.f - ug) * prev;
        p.det[(size_t)m * DETER + col] = nd;
        p.ndbf[(size_t)m * DETER + col] = f2bf(nd);
        p.out[O_DE + ((size_t)m * LL + t) * DETER + col] = nd;
      }
    }
    gbar(p.bar, epoch, NB);

    // ---- G6: obs GEMM, A = [new_deter(4096) | embed_t(1024)], K=5120, split 4 ----
    for (int task = gw; task < 256; task += GW) {
      int ks = task & 3, nt = task >> 2;
      const unsigned* Btile = p.w_obs + (size_t)nt * 160 * 256;
      v8f acc = {};
      for (int kc = ks * 40; kc < ks * 40 + 40; ++kc) {
        const unsigned short* A;
        int lda, k;
        if (kc < 128) { A = p.ndbf; lda = DETER; k = kc * 32; }
        else { A = p.emb_bf + (size_t)t * EMB; lda = LL * EMB; k = (kc - 128) * 32; }
        acc = wmma_bf16(load_a16(A, lda, lane, k), load_b16(Btile, lane, kc), acc);
      }
      int n = nt * 16 + (lane & 15);
      int mb = (lane & 16) ? 8 : 0;
#pragma unroll
      for (int r = 0; r < 8; r++) atomicAdd(&p.opre[(size_t)(mb + r) * HIDDEN + n], acc[r]);
    }
    gbar(p.bar, epoch, NB);

    // ---- E7a: sumsq per row over 1024 ----
    for (int w = gw; w < BB; w += GW) {
      float s = 0.f;
      for (int i = lane; i < HIDDEN; i += 32) {
        float v = p.opre[(size_t)w * HIDDEN + i] + p.obs_b0[i];
        s += v * v;
      }
      s = wave_sum(s);
      if (lane == 0) p.mso[w] = s * (1.f / 1024.f);
    }
    gbar(p.bar, epoch, NB);

    // ---- E7b: rmsnorm + SiLU -> albf ----
    for (int i = tid; i < BB * HIDDEN; i += NT) {
      int b = i >> 10, c = i & 1023;
      float v = (p.opre[i] + p.obs_b0[c]) * rsqrtf(p.mso[b] + 1e-4f) * p.obs_g0[c];
      p.albf[i] = f2bf(siluf(v));
    }
    gbar(p.bar, epoch, NB);

    // ---- G8: logit GEMM (K=1024, direct store + bias, also to output) ----
    for (int task = gw; task < 64; task += GW) {
      const unsigned* Btile = p.w_lw + (size_t)task * 32 * 256;
      v8f acc = {};
      for (int kc = 0; kc < 32; ++kc)
        acc = wmma_bf16(load_a16(p.albf, HIDDEN, lane, kc * 32),
                        load_b16(Btile, lane, kc), acc);
      int n = task * 16 + (lane & 15);
      int mb = (lane & 16) ? 8 : 0;
#pragma unroll
      for (int r = 0; r < 8; r++) {
        int m = mb + r;
        float v = acc[r] + p.obs_lb[n];
        p.logit[(size_t)m * HIDDEN + n] = v;
        p.out[O_LG + ((size_t)m * LL + t) * HIDDEN + n] = v;
      }
    }
    gbar(p.bar, epoch, NB);

    // ---- E9: softmax + unimix + categorical sample (hash-RNG surrogate) ----
    for (int g0 = tid; g0 < BB * STOCH; g0 += NT) {
      int b = g0 >> 5, s = g0 & 31;
      const float* lg = p.logit + (size_t)b * HIDDEN + s * DISC;
      float mx = -1e30f;
      for (int d = 0; d < DISC; d++) mx = fmaxf(mx, lg[d]);
      float pr[DISC], sum = 0.f;
      for (int d = 0; d < DISC; d++) { pr[d] = __expf(lg[d] - mx); sum += pr[d]; }
      float inv = 1.f / sum;
      for (int d = 0; d < DISC; d++) pr[d] = 0.99f * pr[d] * inv + 0.01f / DISC;
      unsigned h = 0x9E3779B9u * (unsigned)(t * 512 + g0) + 0x85EBCA6Bu;
      h ^= h >> 16; h *= 0x7FEB352Du; h ^= h >> 15; h *= 0x846CA68Bu; h ^= h >> 16;
      float u = (float)(h >> 8) * (1.f / 16777216.f);
      int idx = DISC - 1; float cs = 0.f;
      for (int d = 0; d < DISC; d++) { cs += pr[d]; if (u < cs) { idx = d; break; } }
      for (int d = 0; d < DISC; d++) {
        float o = (d == idx) ? 1.f : 0.f;       // straight-through forward value
        p.sto[(size_t)b * FLAT + s * DISC + d] = o;
        p.out[O_ST + ((size_t)b * LL + t) * FLAT + s * DISC + d] = o;
      }
    }
    gbar(p.bar, epoch, NB);
  }
}

// ---------------- host entry ----------------
extern "C" void kernel_launch(void* const* d_in, const int* in_sizes, int n_in,
                              void* d_out, int out_size, void* d_ws, size_t ws_size,
                              hipStream_t stream) {
  (void)in_sizes; (void)n_in; (void)out_size; (void)ws_size;
  // setup_inputs() order: embed, action, reset, d_emb, stoch0, deter0, then params dict
  const float* embed  = (const float*)d_in[0];
  const float* action = (const float*)d_in[1];
  const unsigned char* reset = (const unsigned char*)d_in[2];  // jnp bool = 1 byte
  const float* d_emb  = (const float*)d_in[3];
  const float* stoch0 = (const float*)d_in[4];
  const float* deter0 = (const float*)d_in[5];
  const float* in0_w = (const float*)d_in[6];
  const float* in0_b = (const float*)d_in[7];
  const float* in0_g = (const float*)d_in[8];
  const float* in1_w = (const float*)d_in[9];
  const float* in1_b = (const float*)d_in[10];
  const float* in1_g = (const float*)d_in[11];
  const float* in2_w = (const float*)d_in[12];
  const float* in2_b = (const float*)d_in[13];
  const float* in2_g = (const float*)d_in[14];
  const float* in3_w = (const float*)d_in[15];
  const float* in3_b = (const float*)d_in[16];
  const float* in3_g = (const float*)d_in[17];
  const float* hid_w = (const float*)d_in[18];
  const float* hid_b = (const float*)d_in[19];
  const float* hid_g = (const float*)d_in[20];
  const float* gru_w = (const float*)d_in[21];
  const float* gru_b = (const float*)d_in[22];
  const float* obs_w0 = (const float*)d_in[23];
  const float* obs_b0 = (const float*)d_in[24];
  const float* obs_g0 = (const float*)d_in[25];
  const float* obs_lw = (const float*)d_in[26];
  const float* obs_lb = (const float*)d_in[27];

  char* ws = (char*)d_ws;
  size_t off = 0;
  auto alloc = [&](size_t bytes) -> char* {
    char* q = ws + off;
    off = (off + bytes + 255) & ~(size_t)255;
    return q;
  };
  unsigned* bar    = (unsigned*)alloc(256);
  unsigned* w_in0  = (unsigned*)alloc(PW_IN0 * 4);
  unsigned* w_in1  = (unsigned*)alloc(PW_IN1 * 4);
  unsigned* w_in2  = (unsigned*)alloc(PW_IN2 * 4);
  unsigned* w_in3  = (unsigned*)alloc(PW_IN3 * 4);
  unsigned* w_hid  = (unsigned*)alloc(PW_HID * 4);
  unsigned* w_gru  = (unsigned*)alloc(PW_GRU * 4);
  unsigned* w_obs  = (unsigned*)alloc(PW_OBS * 4);
  unsigned* w_lw   = (unsigned*)alloc(PW_LW * 4);
  unsigned short* emb_bf = (unsigned short*)alloc((size_t)BB * LL * EMB * 2);
  unsigned short* A0   = (unsigned short*)alloc((size_t)BB * DETER * 2);
  unsigned short* A1   = (unsigned short*)alloc((size_t)BB * FLAT * 2);
  unsigned short* A2   = (unsigned short*)alloc((size_t)BB * 32 * 2);
  unsigned short* A3   = (unsigned short*)alloc((size_t)BB * 32 * 2);
  unsigned short* xbf  = (unsigned short*)alloc((size_t)BB * DETER * 2);
  unsigned short* hbf  = (unsigned short*)alloc((size_t)BB * DETER * 2);
  unsigned short* ndbf = (unsigned short*)alloc((size_t)BB * DETER * 2);
  unsigned short* albf = (unsigned short*)alloc((size_t)BB * HIDDEN * 2);
  float* xpre  = (float*)alloc((size_t)BB * DETER * 4);
  float* hpre  = (float*)alloc((size_t)BB * DETER * 4);
  float* opre  = (float*)alloc((size_t)BB * HIDDEN * 4);
  float* logit = (float*)alloc((size_t)BB * HIDDEN * 4);
  float* det   = (float*)alloc((size_t)BB * DETER * 4);
  float* sto   = (float*)alloc((size_t)BB * FLAT * 4);
  float* msx   = (float*)alloc(64 * 4);
  float* msh   = (float*)alloc(16 * 4);
  float* mso   = (float*)alloc(16 * 4);

  // prep: pack all weights into WMMA-B bf16 layouts (HBM read once; then L2-resident)
  pack_bf16_B<<<dim3(512, 1), 256, 0, stream>>>(in0_w, w_in0, 4096, 4096, 1024);
  pack_bf16_B<<<dim3(256, 1), 256, 0, stream>>>(in1_w, w_in1, 1024, 1024, 1024);
  pack_bf16_B<<<dim3(16, 1),  256, 0, stream>>>(in2_w, w_in2, 32, 32, 1024);
  pack_bf16_B<<<dim3(16, 1),  256, 0, stream>>>(in3_w, w_in3, 32, 16, 1024);  // K padded
  pack_bf16_B<<<dim3(256, 8), 256, 0, stream>>>(hid_w, w_hid, 4608, 4608, 512);
  pack_bf16_B<<<dim3(128, 8), 256, 0, stream>>>(gru_w, w_gru, 512, 512, 1536);
  pack_bf16_B<<<dim3(512, 1), 256, 0, stream>>>(obs_w0, w_obs, 5120, 5120, 1024);
  pack_bf16_B<<<dim3(256, 1), 256, 0, stream>>>(obs_lw, w_lw, 1024, 1024, 1024);
  prep_misc<<<256, 256, 0, stream>>>(embed, stoch0, deter0, emb_bf, sto, det, bar);

  RP p;
  p.action = action; p.d_emb = d_emb; p.reset = reset;
  p.in0_b = in0_b; p.in0_g = in0_g; p.in1_b = in1_b; p.in1_g = in1_g;
  p.in2_b = in2_b; p.in2_g = in2_g; p.in3_b = in3_b; p.in3_g = in3_g;
  p.hid_b = hid_b; p.hid_g = hid_g; p.gru_b = gru_b;
  p.obs_b0 = obs_b0; p.obs_g0 = obs_g0; p.obs_lb = obs_lb;
  p.w_in0 = w_in0; p.w_in1 = w_in1; p.w_in2 = w_in2; p.w_in3 = w_in3;
  p.w_hid = w_hid; p.w_gru = w_gru; p.w_obs = w_obs; p.w_lw = w_lw;
  p.emb_bf = emb_bf;
  p.A0 = A0; p.A1 = A1; p.A2 = A2; p.A3 = A3;
  p.xbf = xbf; p.hbf = hbf; p.ndbf = ndbf; p.albf = albf;
  p.xpre = xpre; p.hpre = hpre; p.opre = opre; p.logit = logit;
  p.det = det; p.sto = sto; p.msx = msx; p.msh = msh; p.mso = mso;
  p.bar = bar; p.out = (float*)d_out;

  rssm_scan_persistent<<<NBLK_GRID, THREADS, 0, stream>>>(p);
}
